// MoELayer_32908039422583
// MI455X (gfx1250) — matrix operand order repaired and checked by reading, not compiled, not run on hardware
//
#include <hip/hip_runtime.h>
#include <hip/hip_bf16.h>
#include <stdint.h>

// ---- problem dims ----
#define Bb   4
#define Ss   1024
#define Dd   1024
#define Hh   16
#define Ff   4096
#define Ee   8
#define Kk   2
#define NAa  2
#define Ll   256
#define Nn   (Bb * Ss)        // 4096
#define CAPc 1280             // K*N*5/(E*4)
#define DHd  64

typedef __attribute__((ext_vector_type(16))) __bf16         v16bf;
typedef __attribute__((ext_vector_type(8)))  float          v8f;
typedef __attribute__((ext_vector_type(16))) unsigned short v16u;
typedef __attribute__((ext_vector_type(8)))  unsigned short v8u;
typedef __attribute__((ext_vector_type(4)))  unsigned int   u32x4;
typedef __attribute__((ext_vector_type(8)))  unsigned int   u32x8;

__device__ __forceinline__ __bf16 f2bf(float f) {
  unsigned u = __builtin_bit_cast(unsigned, f);
  u += 0x7FFFu + ((u >> 16) & 1u);
  unsigned short h = (unsigned short)(u >> 16);
  return __builtin_bit_cast(__bf16, h);
}
__device__ __forceinline__ float bf2f(__bf16 b) {
  unsigned short h = __builtin_bit_cast(unsigned short, b);
  unsigned u = ((unsigned)h) << 16;
  return __builtin_bit_cast(float, u);
}
__device__ __forceinline__ float gelu_f(float x) {
  return 0.5f * x * (1.0f + tanhf(0.7978845608f * (x + 0.044715f * x * x * x)));
}
__device__ __forceinline__ v8f wmma_bf16(v16bf a, v16bf b, v8f c) {
  // (neg_a, A, neg_b, B, c_mod, C, reuse_a, reuse_b)
  return __builtin_amdgcn_wmma_f32_16x16x32_bf16(false, a, false, b, (short)0, c,
                                                 false, false);
}

// A fragment (16x32 bf16). Lane m = lane&15, hf = lane>>4.
// Layout: element j -> k = hf*16 + j  => two contiguous 16B chunks.
__device__ __forceinline__ v16bf load_a_frag(const __bf16* rowptr, int hf) {
  v8u a = *(const v8u*)(rowptr + hf * 16);
  v8u c = *(const v8u*)(rowptr + hf * 16 + 8);
  v16u r;
#pragma unroll
  for (int j = 0; j < 8; ++j) { r[j] = a[j]; r[8 + j] = c[j]; }
  return __builtin_bit_cast(v16bf, r);
}
// B fragment (32x16 bf16). Lane col n = lane&15, k = hf*16 + j (contiguous 32B in
// K-major storage). Caller folds n*ldb + hf*16 into the pointer.
__device__ __forceinline__ v16bf load_b_frag(const __bf16* p) {
  return __builtin_bit_cast(v16bf, *(const v16u*)p);
}

// Async global->LDS copy of one 16B chunk per lane (ASYNCcnt-tracked, gfx1250).
__device__ __forceinline__ void stage_a_async(const __bf16* gsrc, __bf16* ldst) {
  unsigned l = (unsigned)(uintptr_t)ldst;                 // LDS byte address
  unsigned long long g = (unsigned long long)(uintptr_t)gsrc;
  asm volatile("global_load_async_to_lds_b128 %0, %1, off"
               :: "v"(l), "v"(g) : "memory");
}

// ---------------- generic WMMA GEMM ----------------
// A: [M,K] bf16 row-major (lda). Bt: [N,K] bf16 K-major (ldb). C: [M,N] (ldc).
// Block: 256 thr = 8 waves = 2(M) x 4(N); block tile 32 x 256.
// A tile (32x64) is double-buffered in LDS via async-to-LDS DMA.
template <bool OUT_BF16, int RES, bool GELU>
__global__ __launch_bounds__(256) void wmma_gemm_kernel(
    const __bf16* __restrict__ A, int lda, const __bf16* __restrict__ Bt, int ldb,
    void* __restrict__ Cout, int ldc, const void* __restrict__ Res, int ldr, int Ktot) {
  __shared__ __attribute__((aligned(16))) __bf16 As[2][32 * 64];
  const int tid = threadIdx.x;
  const int lane = tid & 31, wave = tid >> 5;
  const int wm = wave >> 2, wn = wave & 3;
  const int m = lane & 15, hf = lane >> 4;
  const int m0 = blockIdx.y * 32;
  const int n0 = blockIdx.x * 256 + wn * 64;

  const int srow = tid >> 3;       // 0..31
  const int scol = (tid & 7) * 8;  // 8 chunks of 8 elems (16B) -> 64 cols
  const __bf16* aSrc = A + (size_t)(m0 + srow) * lda + scol;
  __bf16* aDst0 = &As[0][srow * 64 + scol];
  __bf16* aDst1 = &As[1][srow * 64 + scol];

  v8f acc[4] = {};
  const __bf16* bBase[4];
#pragma unroll
  for (int nt = 0; nt < 4; ++nt)
    bBase[nt] = Bt + (size_t)(n0 + nt * 16 + m) * ldb + hf * 16;

  stage_a_async(aSrc, aDst0);
  const int nkt = Ktot >> 6;
  for (int t = 0; t < nkt; ++t) {
    const int k0 = t << 6;
    if (t + 1 < nkt) {
      stage_a_async(aSrc + k0 + 64, ((t + 1) & 1) ? aDst1 : aDst0);
      asm volatile("s_wait_asynccnt 0x1" ::: "memory");
    } else {
      asm volatile("s_wait_asynccnt 0x0" ::: "memory");
    }
    __syncthreads();
    const __bf16* abase = &As[t & 1][(wm * 16 + m) * 64];
#pragma unroll
    for (int kk = 0; kk < 64; kk += 32) {
      v16bf aF = load_a_frag(abase + kk, hf);
#pragma unroll
      for (int nt = 0; nt < 4; ++nt) {
        v16bf bF = load_b_frag(bBase[nt] + k0 + kk);
        __builtin_prefetch((const void*)(bBase[nt] + k0 + 64), 0, 1);
        acc[nt] = wmma_bf16(aF, bF, acc[nt]);
      }
    }
    __syncthreads();
  }
#pragma unroll
  for (int nt = 0; nt < 4; ++nt) {
#pragma unroll
    for (int i = 0; i < 8; ++i) {
      const int row = m0 + wm * 16 + hf * 8 + i;
      const int col = n0 + nt * 16 + m;
      float v = acc[nt][i];
      if (RES == 1) v += ((const float*)Res)[(size_t)row * ldr + col];
      if (RES == 2) v += bf2f(((const __bf16*)Res)[(size_t)row * ldr + col]);
      if (GELU) v = gelu_f(v);
      if (OUT_BF16) ((__bf16*)Cout)[(size_t)row * ldc + col] = f2bf(v);
      else          ((float*)Cout)[(size_t)row * ldc + col] = v;
    }
  }
}

// ---------------- layernorm -> bf16 ----------------
__global__ __launch_bounds__(256) void ln_kernel(const float* __restrict__ x,
                                                 const float* __restrict__ g,
                                                 const float* __restrict__ be,
                                                 __bf16* __restrict__ h) {
  __shared__ float red[256];
  const int row = blockIdx.x, t = threadIdx.x;
  const float* xr = x + (size_t)row * Dd;
  float s = 0.f;
  for (int d = t; d < Dd; d += 256) s += xr[d];
  red[t] = s; __syncthreads();
  for (int o = 128; o > 0; o >>= 1) { if (t < o) red[t] += red[t + o]; __syncthreads(); }
  const float mean = red[0] * (1.0f / Dd);
  __syncthreads();
  float v = 0.f;
  for (int d = t; d < Dd; d += 256) { float df = xr[d] - mean; v += df * df; }
  red[t] = v; __syncthreads();
  for (int o = 128; o > 0; o >>= 1) { if (t < o) red[t] += red[t + o]; __syncthreads(); }
  const float rs = rsqrtf(red[0] * (1.0f / Dd) + 1e-5f);
  for (int d = t; d < Dd; d += 256)
    h[(size_t)row * Dd + d] = f2bf((xr[d] - mean) * rs * g[d] + be[d]);
}

// ---------------- weight transpose+convert: W[Kd,Nd] f32 -> Wt[Nd,Kd] bf16 -----
__global__ void conv_transpose_kernel(const float* __restrict__ W,
                                      __bf16* __restrict__ Wt, int Kd, int Nd) {
  __shared__ float tile[32][33];
  const int bat = blockIdx.z;
  const float* Wb = W + (size_t)bat * Kd * Nd;
  __bf16* Wtb = Wt + (size_t)bat * Kd * Nd;
  const int n0 = blockIdx.x * 32, k0 = blockIdx.y * 32;
  const int tx = threadIdx.x, ty = threadIdx.y;
  for (int r = ty; r < 32; r += 8) tile[r][tx] = Wb[(size_t)(k0 + r) * Nd + n0 + tx];
  __syncthreads();
  for (int r = ty; r < 32; r += 8)
    Wtb[(size_t)(n0 + r) * Kd + k0 + tx] = f2bf(tile[tx][r]);
}

__global__ void conv_bf_kernel(const float* __restrict__ in, __bf16* __restrict__ out,
                               int n) {
  int i = blockIdx.x * 256 + threadIdx.x;
  if (i < n) out[i] = f2bf(in[i]);
}

// ---------------- flash self-attention (1 wave per (b,h,16-q tile)) ------------
// V chunks (32 keys x 64 dims, row stride 3*D) are staged into LDS by the
// Tensor Data Mover (TENSORcnt-tracked); EXEC is ignored for TDM and the block
// is a single wave, so each issue is exactly one DMA.
__global__ __launch_bounds__(32) void attn_kernel(const __bf16* __restrict__ qkv,
                                                  const unsigned char* __restrict__ pad,
                                                  __bf16* __restrict__ ctx) {
  __shared__ __attribute__((aligned(16))) __bf16 pls[16 * 32];
  __shared__ __attribute__((aligned(16))) __bf16 vls[32 * DHd];
  const int q0 = blockIdx.x * 16, h = blockIdx.y, b = blockIdx.z;
  const int lane = threadIdx.x;
  const int m = lane & 15, hf = lane >> 4;
  const int ld = 3 * Dd;
  const size_t rowbase = (size_t)b * Ss;

  v16bf qf[2];
  {
    const __bf16* qrow = qkv + (rowbase + q0 + m) * ld + h * DHd;
#pragma unroll
    for (int kc = 0; kc < 2; ++kc) qf[kc] = load_a_frag(qrow + kc * 32, hf);
  }
  v8f o[4] = {};
  float rmax[8], rsum[8];
#pragma unroll
  for (int i = 0; i < 8; ++i) { rmax[i] = -1e30f; rsum[i] = 0.f; }

  const unsigned vlsOff = (unsigned)(uintptr_t)&vls[0];
  const int nch = (q0 + 47) >> 5;
  for (int ch = 0; ch < nch; ++ch) {
    const int kb = ch * 32;
    {  // TDM: 2D tile, data_size=2B, tile 64x32, tensor_dim0_stride = 3*D
      unsigned long long ga = (unsigned long long)(uintptr_t)(
          qkv + (rowbase + kb) * ld + 2 * Dd + h * DHd);
      u32x4 g0 = { 1u,                                   // count=1
                   vlsOff,                               // lds_addr [63:32]
                   (unsigned)ga,                         // global_addr lo
                   (unsigned)(ga >> 32) | 0x80000000u }; // addr hi | type=2<<30
      u32x8 g1 = { 0x00010000u,         // data_size=1 (2 bytes), mask=0
                   64u << 16,           // tensor_dim0 = 64  (bits 79:48 lo16)
                   32u << 16,           // tensor_dim1 = 32  (bits 111:80 lo16)
                   64u << 16,           // tile_dim0  = 64   (bits 127:112)
                   32u,                 // tile_dim1  = 32   (bits 143:128)
                   (unsigned)(3 * Dd),  // tensor_dim0_stride lo32
                   0u, 0u };
      asm volatile("tensor_load_to_lds %0, %1" :: "s"(g0), "s"(g1) : "memory");
    }
    v8f sc0 = {}, sc1 = {};
#pragma unroll
    for (int kc = 0; kc < 2; ++kc) {
      const __bf16* kr0 = qkv + (rowbase + kb + m) * ld + Dd + h * DHd + kc * 32 + hf * 16;
      const __bf16* kr1 = qkv + (rowbase + kb + 16 + m) * ld + Dd + h * DHd + kc * 32 + hf * 16;
      sc0 = wmma_bf16(qf[kc], load_b_frag(kr0), sc0);
      sc1 = wmma_bf16(qf[kc], load_b_frag(kr1), sc1);
    }
    float fs[8];
#pragma unroll
    for (int i = 0; i < 8; ++i) {
      const int qrow = q0 + hf * 8 + i;
      const int kA = kb + m, kB = kb + 16 + m;
      float s0 = (kA <= qrow && pad[b * Ss + kA]) ? sc0[i] * 0.125f : -1e9f;
      float s1 = (kB <= qrow && pad[b * Ss + kB]) ? sc1[i] * 0.125f : -1e9f;
      float mx = fmaxf(s0, s1);
#pragma unroll
      for (int d = 1; d < 16; d <<= 1) mx = fmaxf(mx, __shfl_xor(mx, d, 32));
      const float nm = fmaxf(rmax[i], mx);
      const float f = __expf(rmax[i] - nm);
      const float e0 = __expf(s0 - nm), e1 = __expf(s1 - nm);
      float sm = e0 + e1;
#pragma unroll
      for (int d = 1; d < 16; d <<= 1) sm += __shfl_xor(sm, d, 32);
      rsum[i] = rsum[i] * f + sm;
      rmax[i] = nm;
      fs[i] = f;
      pls[(hf * 8 + i) * 32 + m] = f2bf(e0);
      pls[(hf * 8 + i) * 32 + 16 + m] = f2bf(e1);
    }
#pragma unroll
    for (int t = 0; t < 4; ++t)
#pragma unroll
      for (int i = 0; i < 8; ++i) o[t][i] *= fs[i];
    __builtin_amdgcn_s_wait_tensorcnt(0);  // V tile landed in LDS
    __syncthreads();
    v16bf pf = load_a_frag(&pls[m * 32], hf);
    const unsigned short* vu = (const unsigned short*)vls;
#pragma unroll
    for (int t = 0; t < 4; ++t) {
      v16u bv;
#pragma unroll
      for (int j = 0; j < 16; ++j) bv[j] = vu[(hf * 16 + j) * DHd + t * 16 + m];
      o[t] = wmma_bf16(pf, __builtin_bit_cast(v16bf, bv), o[t]);
    }
    __syncthreads();
  }
#pragma unroll
  for (int t = 0; t < 4; ++t)
#pragma unroll
    for (int i = 0; i < 8; ++i) {
      const size_t row = rowbase + q0 + hf * 8 + i;
      ctx[row * Dd + h * DHd + t * 16 + m] = f2bf(o[t][i] / rsum[i]);
    }
}

// ---------------- router: logits -> softmax -> top-2 ----------------
__global__ __launch_bounds__(256) void router_kernel(const float* __restrict__ x1,
                                                     const int* __restrict__ avail,
                                                     const float* __restrict__ wr,
                                                     const float* __restrict__ br,
                                                     int* __restrict__ idx_top,
                                                     float* __restrict__ w_top) {
  const int n = blockIdx.x * 256 + threadIdx.x;
  if (n >= Nn) return;
  const float* xr = x1 + (size_t)n * Dd;
  const float av = (float)avail[n / Ss];
  float lg[Ee];
#pragma unroll
  for (int e = 0; e < Ee; ++e) lg[e] = br[e];
  for (int d = 0; d < Dd; ++d) {
    const float xv = xr[d];
#pragma unroll
    for (int e = 0; e < Ee; ++e) lg[e] += xv * wr[d * Ee + e];
  }
#pragma unroll
  for (int e = 0; e < Ee; ++e) lg[e] += av * wr[Dd * Ee + e];
  float mx = lg[0];
#pragma unroll
  for (int e = 1; e < Ee; ++e) mx = fmaxf(mx, lg[e]);
  float p[Ee], sum = 0.f;
#pragma unroll
  for (int e = 0; e < Ee; ++e) { p[e] = __expf(lg[e] - mx); sum += p[e]; }
  const float inv = 1.0f / sum;
#pragma unroll
  for (int e = 0; e < Ee; ++e) p[e] *= inv;
  int i0 = 0;
#pragma unroll
  for (int e = 1; e < Ee; ++e) if (p[e] > p[i0]) i0 = e;
  int i1 = (i0 == 0) ? 1 : 0;
#pragma unroll
  for (int e = 0; e < Ee; ++e) if (e != i0 && p[e] > p[i1]) i1 = e;
  idx_top[n * Kk + 0] = i0;  idx_top[n * Kk + 1] = i1;
  w_top[n * Kk + 0] = p[i0]; w_top[n * Kk + 1] = p[i1];
}

// ---------------- deterministic capacity rank scan (single thread) -------------
__global__ void rank_kernel(const int* __restrict__ idx_top,
                            const float* __restrict__ w_top, int* __restrict__ rcg,
                            float* __restrict__ wflat, int* __restrict__ slot_tok,
                            int* __restrict__ bidx) {
  if (threadIdx.x != 0 || blockIdx.x != 0) return;
  int cnt[Ee];
#pragma unroll
  for (int e = 0; e < Ee; ++e) cnt[e] = 0;
  for (int i = 0; i < Kk * Nn; ++i) {
    const int k = i / Nn, n = i % Nn;
    const int e = idx_top[n * Kk + k];
    const int r = cnt[e]++;
    const bool valid = r < CAPc;
    rcg[i] = valid ? r : (CAPc - 1);
    wflat[i] = valid ? w_top[n * Kk + k] : 0.f;
    if (valid) { slot_tok[e * CAPc + r] = n; bidx[e * CAPc + r] = n / Ss; }
  }
}

__global__ __launch_bounds__(256) void dispatch_kernel(const int* __restrict__ slot_tok,
                                                       const float* __restrict__ x1,
                                                       __bf16* __restrict__ disp) {
  const int slot = blockIdx.x;
  const int tok = slot_tok[slot];
  if (tok < 0) return;
  for (int d = threadIdx.x; d < Dd; d += 256)
    disp[(size_t)slot * Dd + d] = f2bf(x1[(size_t)tok * Dd + d]);
}

// ---------------- cross-attention for expert-A slots (scalar path) -------------
__global__ __launch_bounds__(64) void xattn_kernel(const float* __restrict__ qa,
                                                   const float* __restrict__ ka,
                                                   const float* __restrict__ va,
                                                   const int* __restrict__ bidx,
                                                   const unsigned char* __restrict__ em,
                                                   __bf16* __restrict__ oa) {
  __shared__ float p[Ll];
  const int slot = blockIdx.x, h = blockIdx.y, e = blockIdx.z;
  const int t = threadIdx.x;
  const int gslot = e * CAPc + slot;
  const int b = bidx[gslot];
  const float* qv = qa + (size_t)gslot * Dd + h * DHd;
  const float* kb = ka + ((size_t)e * Bb * Ll + (size_t)b * Ll) * Dd + h * DHd;
  const float* vb = va + ((size_t)e * Bb * Ll + (size_t)b * Ll) * Dd + h * DHd;
  for (int l = t * 4; l < t * 4 + 4; ++l) {
    float s = -1e9f;
    if (em[b * Ll + l]) {
      float acc = 0.f;
      for (int d = 0; d < DHd; ++d) acc += qv[d] * kb[(size_t)l * Dd + d];
      s = acc * 0.125f;
    }
    p[l] = s;
  }
  __syncthreads();
  float mx = -1e30f;
  for (int l = 0; l < Ll; ++l) mx = fmaxf(mx, p[l]);
  __syncthreads();
  for (int l = t * 4; l < t * 4 + 4; ++l) p[l] = __expf(p[l] - mx);
  __syncthreads();
  float sum = 0.f;
  for (int l = 0; l < Ll; ++l) sum += p[l];
  const float inv = 1.0f / sum;
  float od = 0.f;
  for (int l = 0; l < Ll; ++l) od += p[l] * vb[(size_t)l * Dd + t];
  oa[(size_t)gslot * Dd + h * DHd + t] = f2bf(od * inv);
}

// ---------------- final combine: out = x1 + sum_k w * eout[e, slot] ------------
__global__ __launch_bounds__(256) void combine_kernel(const float* __restrict__ x1,
                                                      const float* __restrict__ eout,
                                                      const int* __restrict__ idx_top,
                                                      const int* __restrict__ rcg,
                                                      const float* __restrict__ wflat,
                                                      float* __restrict__ out) {
  const int n = blockIdx.x;
  const float w0 = wflat[0 * Nn + n], w1 = wflat[1 * Nn + n];
  const int e0 = idx_top[n * Kk + 0], e1 = idx_top[n * Kk + 1];
  const int r0 = rcg[0 * Nn + n], r1 = rcg[1 * Nn + n];
  const float* g0 = eout + ((size_t)e0 * CAPc + r0) * Dd;
  const float* g1 = eout + ((size_t)e1 * CAPc + r1) * Dd;
  for (int d = threadIdx.x; d < Dd; d += 256)
    out[(size_t)n * Dd + d] = x1[(size_t)n * Dd + d] + w0 * g0[d] + w1 * g1[d];
}

// =========================== host launcher ===========================
extern "C" void kernel_launch(void* const* d_in, const int* in_sizes, int n_in,
                              void* d_out, int out_size, void* d_ws, size_t ws_size,
                              hipStream_t stream) {
  (void)in_sizes; (void)n_in; (void)out_size; (void)ws_size;
  const float* x       = (const float*)d_in[0];
  const float* enc     = (const float*)d_in[1];
  const unsigned char* emask = (const unsigned char*)d_in[2];
  const unsigned char* pmask = (const unsigned char*)d_in[3];
  const int*   avail   = (const int*)d_in[4];
  const float* ln_g    = (const float*)d_in[5];
  const float* ln_b    = (const float*)d_in[6];
  const float* w_qkv   = (const float*)d_in[7];
  const float* w_o     = (const float*)d_in[8];
  const float* w_rt    = (const float*)d_in[9];
  const float* b_rt    = (const float*)d_in[10];
  const float* a_wq    = (const float*)d_in[11];
  const float* a_wk    = (const float*)d_in[12];
  const float* a_wv    = (const float*)d_in[13];
  const float* a_wo    = (const float*)d_in[14];
  const float* a_w1    = (const float*)d_in[15];
  const float* a_w2    = (const float*)d_in[16];
  const float* b_w1    = (const float*)d_in[17];
  const float* b_w2    = (const float*)d_in[18];
  float* out = (float*)d_out;

  char* ws = (char*)d_ws;
  size_t off = 0;
  auto alloc = [&](size_t bytes) -> char* {
    char* p = ws + off;
    off += (bytes + 255) & ~(size_t)255;
    return p;
  };
  __bf16* wqkvT = (__bf16*)alloc((size_t)3 * Dd * Dd * 2);
  __bf16* woT   = (__bf16*)alloc((size_t)Dd * Dd * 2);
  __bf16* awqT  = (__bf16*)alloc((size_t)NAa * Dd * Dd * 2);
  __bf16* awkT  = (__bf16*)alloc((size_t)NAa * Dd * Dd * 2);
  __bf16* awvT  = (__bf16*)alloc((size_t)NAa * Dd * Dd * 2);
  __bf16* awoT  = (__bf16*)alloc((size_t)NAa * Dd * Dd * 2);
  __bf16* aw1T  = (__bf16*)alloc((size_t)NAa * Dd * Ff * 2);
  __bf16* aw2T  = (__bf16*)alloc((size_t)NAa * Dd * Ff * 2);
  __bf16* bw1T  = (__bf16*)alloc((size_t)(Ee - NAa) * Dd * Ff * 2);
  __bf16* bw2T  = (__bf16*)alloc((size_t)(Ee - NAa) * Dd * Ff * 2);
  __bf16* hbf   = (__bf16*)alloc((size_t)Nn * Dd * 2);
  __bf16* qkvbf = (__bf16*)alloc((size_t)Nn * 3 * Dd * 2);
  __bf16* ctxbf = (__bf16*)alloc((size_t)Nn * Dd * 2);
  float*  x1    = (float*)alloc((size_t)Nn * Dd * 4);
  __bf16* encbf = (__bf16*)alloc((size_t)Bb * Ll * Dd * 2);
  int*    idxtop= (int*)alloc((size_t)Nn * Kk * 4);
  float*  wtop  = (float*)alloc((size_t)Nn * Kk * 4);
  int*    rcg   = (int*)alloc((size_t)Nn * Kk * 4);
  float*  wflat = (float*)alloc((size_t)Nn * Kk * 4);
  int*    slottok=(int*)alloc((size_t)Ee * CAPc * 4);
  int*    bidx  = (int*)alloc((size_t)Ee * CAPc * 4);
  __bf16* disp  = (__bf16*)alloc((size_t)Ee * CAPc * Dd * 2);
  float*  qa    = (float*)alloc((size_t)NAa * CAPc * Dd * 4);
  float*  ka    = (float*)alloc((size_t)NAa * Bb * Ll * Dd * 4);
  float*  va    = (float*)alloc((size_t)NAa * Bb * Ll * Dd * 4);
  __bf16* oabf  = (__bf16*)alloc((size_t)NAa * CAPc * Dd * 2);
  __bf16* habf  = (__bf16*)alloc((size_t)NAa * CAPc * Dd * 2);
  __bf16* tbuf  = (__bf16*)alloc((size_t)CAPc * Ff * 2);
  float*  eout  = (float*)alloc((size_t)Ee * CAPc * Dd * 4);

  (void)hipMemsetAsync(disp, 0, (size_t)Ee * CAPc * Dd * 2, stream);
  (void)hipMemsetAsync(bidx, 0, (size_t)Ee * CAPc * 4, stream);
  (void)hipMemsetAsync(slottok, 0xFF, (size_t)Ee * CAPc * 4, stream);

  // ---- weight conversion (fp32 -> bf16, transposed to K-major) ----
  const dim3 cb(32, 8);
  conv_transpose_kernel<<<dim3(3 * Dd / 32, Dd / 32, 1), cb, 0, stream>>>(w_qkv, wqkvT, Dd, 3 * Dd);
  conv_transpose_kernel<<<dim3(Dd / 32, Dd / 32, 1), cb, 0, stream>>>(w_o, woT, Dd, Dd);
  conv_transpose_kernel<<<dim3(Dd / 32, Dd / 32, NAa), cb, 0, stream>>>(a_wq, awqT, Dd, Dd);
  conv_transpose_kernel<<<dim3(Dd / 32, Dd / 32, NAa), cb, 0, stream>>>(a_wk, awkT, Dd, Dd);
  conv_transpose_kernel<<<dim3(Dd / 32, Dd / 32, NAa), cb, 0, stream>>>(a_wv, awvT, Dd, Dd);
  conv_transpose_kernel<<<dim3(Dd / 32, Dd / 32, NAa), cb, 0, stream>>>(a_wo, awoT, Dd, Dd);
  conv_transpose_kernel<<<dim3(Ff / 32, Dd / 32, NAa), cb, 0, stream>>>(a_w1, aw1T, Dd, Ff);
  conv_transpose_kernel<<<dim3(Dd / 32, Ff / 32, NAa), cb, 0, stream>>>(a_w2, aw2T, Ff, Dd);
  conv_transpose_kernel<<<dim3(Ff / 32, Dd / 32, Ee - NAa), cb, 0, stream>>>(b_w1, bw1T, Dd, Ff);
  conv_transpose_kernel<<<dim3(Dd / 32, Ff / 32, Ee - NAa), cb, 0, stream>>>(b_w2, bw2T, Ff, Dd);
  conv_bf_kernel<<<(Bb * Ll * Dd) / 256, 256, 0, stream>>>(enc, encbf, Bb * Ll * Dd);

  // ---- LN + QKV ----
  ln_kernel<<<Nn, 256, 0, stream>>>(x, ln_g, ln_b, hbf);
  wmma_gemm_kernel<true, 0, false><<<dim3(3 * Dd / 256, Nn / 32), 256, 0, stream>>>(
      hbf, Dd, wqkvT, Dd, qkvbf, 3 * Dd, nullptr, 0, Dd);

  // ---- causal flash attention + output projection (+ residual x) ----
  attn_kernel<<<dim3(Ss / 16, Hh, Bb), 32, 0, stream>>>(qkvbf, pmask, ctxbf);
  wmma_gemm_kernel<false, 1, false><<<dim3(Dd / 256, Nn / 32), 256, 0, stream>>>(
      ctxbf, Dd, woT, Dd, x1, Dd, x, Dd, Dd);

  // ---- router / top-2 / capacity rank / dispatch ----
  router_kernel<<<Nn / 256, 256, 0, stream>>>(x1, avail, w_rt, b_rt, idxtop, wtop);
  rank_kernel<<<1, 32, 0, stream>>>(idxtop, wtop, rcg, wflat, slottok, bidx);
  dispatch_kernel<<<Ee * CAPc, 256, 0, stream>>>(slottok, x1, disp);

  // ---- expert-A cross attention ----
  for (int e = 0; e < NAa; ++e) {
    wmma_gemm_kernel<false, 0, false><<<dim3(Dd / 256, CAPc / 32), 256, 0, stream>>>(
        disp + (size_t)e * CAPc * Dd, Dd, awqT + (size_t)e * Dd * Dd, Dd,
        qa + (size_t)e * CAPc * Dd, Dd, nullptr, 0, Dd);
    wmma_gemm_kernel<false, 0, false><<<dim3(Dd / 256, (Bb * Ll) / 32), 256, 0, stream>>>(
        encbf, Dd, awkT + (size_t)e * Dd * Dd, Dd,
        ka + (size_t)e * Bb * Ll * Dd, Dd, nullptr, 0, Dd);
    wmma_gemm_kernel<false, 0, false><<<dim3(Dd / 256, (Bb * Ll) / 32), 256, 0, stream>>>(
        encbf, Dd, awvT + (size_t)e * Dd * Dd, Dd,
        va + (size_t)e * Bb * Ll * Dd, Dd, nullptr, 0, Dd);
  }
  xattn_kernel<<<dim3(CAPc, Hh, NAa), 64, 0, stream>>>(qa, ka, va, bidx, emask, oabf);
  for (int e = 0; e < NAa; ++e) {
    // ha = xa + oa @ a_wo  (residual from dispatched bf16 rows)
    wmma_gemm_kernel<true, 2, false><<<dim3(Dd / 256, CAPc / 32), 256, 0, stream>>>(
        oabf + (size_t)e * CAPc * Dd, Dd, awoT + (size_t)e * Dd * Dd, Dd,
        habf + (size_t)e * CAPc * Dd, Dd, disp + (size_t)e * CAPc * Dd, Dd, Dd);
    // FFN: t = gelu(ha @ w1); eout = t @ w2
    wmma_gemm_kernel<true, 0, true><<<dim3(Ff / 256, CAPc / 32), 256, 0, stream>>>(
        habf + (size_t)e * CAPc * Dd, Dd, aw1T + (size_t)e * Dd * Ff, Dd,
        tbuf, Ff, nullptr, 0, Dd);
    wmma_gemm_kernel<false, 0, false><<<dim3(Dd / 256, CAPc / 32), 256, 0, stream>>>(
        tbuf, Ff, aw2T + (size_t)e * Dd * Ff, Ff,
        eout + (size_t)e * CAPc * Dd, Dd, nullptr, 0, Ff);
  }
  // ---- expert-B FFNs ----
  for (int i = 0; i < Ee - NAa; ++i) {
    const int e = NAa + i;
    wmma_gemm_kernel<true, 0, true><<<dim3(Ff / 256, CAPc / 32), 256, 0, stream>>>(
        disp + (size_t)e * CAPc * Dd, Dd, bw1T + (size_t)i * Dd * Ff, Dd,
        tbuf, Ff, nullptr, 0, Dd);
    wmma_gemm_kernel<false, 0, false><<<dim3(Dd / 256, CAPc / 32), 256, 0, stream>>>(
        tbuf, Ff, bw2T + (size_t)i * Dd * Ff, Ff,
        eout + (size_t)e * CAPc * Dd, Dd, nullptr, 0, Ff);
  }

  // ---- gather + weighted combine + residual ----
  combine_kernel<<<Nn, 256, 0, stream>>>(x1, eout, idxtop, rcg, wflat, out);
}
